// InformerAutoformerHybrid_43190191129173
// MI455X (gfx1250) — matrix operand order repaired and checked by reading, not compiled, not run on hardware
//
#include <hip/hip_runtime.h>
#include <hip/hip_bf16.h>

// ---------------------------------------------------------------------------
// InformerAutoformerHybrid forward for MI455X (gfx1250, wave32, WMMA).
//
//  * adaptive_decomp is an identity for the final output -> skipped.
//  * Big GEMMs use v_wmma_f32_16x16x32_bf16 with a bf16x3 hi/lo split of both
//    operands (3 WMMA products, f32 accumulate) for near-fp32 accuracy on the
//    bf16 matrix pipe.
//  * LDS layouts are arranged so every WMMA fragment is two contiguous,
//    16B-aligned runs -> fragment assembly is pure ds_load_b128 (no scalar
//    u16 loads / b16 moves). Weights are pre-split *transposed* so B-fragment
//    K-runs are contiguous.
// ---------------------------------------------------------------------------

typedef __attribute__((ext_vector_type(16))) __bf16 v16bf;
typedef __attribute__((ext_vector_type(8)))  __bf16 v8bf;
typedef __attribute__((ext_vector_type(4)))  __bf16 v4bf;
typedef __attribute__((ext_vector_type(8)))  float  v8f;

#define BATCH 16
#define SEQL  4096
#define DIN   8
#define DMODEL 512
#define DFF   2048
#define NHEAD 8
#define DK    64
#define MROWS (BATCH * SEQL)          // 65536
#define OUTL  96
#define UMAX  10

// ---- GEMM tiling ----------------------------------------------------------
#define BM 128
#define BN 128
#define BK 32
#define SA 40                          // LDS row stride (bf16) for A tile
#define SB 40                          // LDS row stride (bf16) for Wt tile

// ---------------------------------------------------------------------------
// Split fp32 weight W[K x N] into TRANSPOSED bf16 hi/lo Wt[N x K].
// ---------------------------------------------------------------------------
__global__ void split_wt_kernel(const float* __restrict__ w,
                                __bf16* __restrict__ hi, __bf16* __restrict__ lo,
                                int K, int N) {
    int i = blockIdx.x * 256 + threadIdx.x;      // over N*K
    if (i < N * K) {
        int n = i / K, k = i - n * K;
        float v = w[(size_t)k * N + n];
        __bf16 h = (__bf16)v;
        hi[i] = h;
        lo[i] = (__bf16)(v - (float)h);
    }
}

// ---------------------------------------------------------------------------
// Embedding: h = x @ w_embed + b_embed   (K = 8, scalar kernel)
// ---------------------------------------------------------------------------
__global__ __launch_bounds__(256) void embed_kernel(const float* __restrict__ x,
                                                    const float* __restrict__ we,
                                                    const float* __restrict__ be,
                                                    float* __restrict__ h) {
    size_t idx = (size_t)blockIdx.x * 256 + threadIdx.x;   // < MROWS*DMODEL
    int    d   = (int)(idx & (DMODEL - 1));
    size_t m   = idx >> 9;
    const float* xr = x + m * DIN;
    float s = be[d];
#pragma unroll
    for (int k = 0; k < DIN; ++k) s += xr[k] * we[k * DMODEL + d];
    h[idx] = s;
}

// ---------------------------------------------------------------------------
// bf16x3 GEMM:  C[M x N] = A[M x K] @ W[K x N] (+bias)(+relu)
// A fp32 (split to bf16 hi/lo while staging); W pre-split transposed bf16.
// 256 threads = 8 waves; macro-tile 128x128; wave tile 32x64 (2x4 WMMA tiles);
// 24 v_wmma per K-step per wave, all fragments via ds_load_b128.
// ---------------------------------------------------------------------------
__global__ __launch_bounds__(256) void gemm_bf16x3_kernel(
    const float* __restrict__ A,
    const __bf16* __restrict__ Wth, const __bf16* __restrict__ Wtl,
    const float* __restrict__ bias,
    float* __restrict__ C,
    int N, int K, int relu) {

    __shared__ __bf16 sAh[BM * SA];
    __shared__ __bf16 sAl[BM * SA];
    __shared__ __bf16 sBh[BN * SB];   // [n][k] : row n holds k0..k0+31
    __shared__ __bf16 sBl[BN * SB];

    const int tid  = threadIdx.x;
    const int wave = tid >> 5;
    const int lane = tid & 31;
    const int hl   = lane >> 4;        // lane half (0/1)
    const int lid  = lane & 15;
    const int wr   = (wave & 3) * 32;  // wave row offset within macro tile
    const int wc   = (wave >> 2) * 64; // wave col offset within macro tile

    const int m0 = blockIdx.y * BM;
    const int n0 = blockIdx.x * BN;

    v8f acc[2][4];
#pragma unroll
    for (int mt = 0; mt < 2; ++mt)
#pragma unroll
        for (int nt = 0; nt < 4; ++nt)
#pragma unroll
            for (int r = 0; r < 8; ++r) acc[mt][nt][r] = 0.0f;

    for (int k0 = 0; k0 < K; k0 += BK) {
        // ---- stage A tile (128x32 fp32 -> bf16 hi/lo), float4 loads -------
#pragma unroll
        for (int i = 0; i < (BM * BK) / (256 * 4); ++i) {   // 4
            int idx4 = tid + i * 256;
            int r = idx4 >> 3, c4 = (idx4 & 7) << 2;
            const float4 v = *(const float4*)&A[(size_t)(m0 + r) * K + (k0 + c4)];
            v4bf h4, l4;
            {
                __bf16 h;
                h = (__bf16)v.x; h4[0] = h; l4[0] = (__bf16)(v.x - (float)h);
                h = (__bf16)v.y; h4[1] = h; l4[1] = (__bf16)(v.y - (float)h);
                h = (__bf16)v.z; h4[2] = h; l4[2] = (__bf16)(v.z - (float)h);
                h = (__bf16)v.w; h4[3] = h; l4[3] = (__bf16)(v.w - (float)h);
            }
            *(v4bf*)&sAh[r * SA + c4] = h4;
            *(v4bf*)&sAl[r * SA + c4] = l4;
        }
        // ---- stage Wt tile (128 n-rows x 32 k), 16B bf16 loads ------------
#pragma unroll
        for (int i = 0; i < (BN * BK) / (256 * 8); ++i) {   // 2
            int idx8 = tid + i * 256;
            int r = idx8 >> 2, c8 = (idx8 & 3) << 3;
            size_t g = (size_t)(n0 + r) * K + (k0 + c8);
            *(v8bf*)&sBh[r * SB + c8] = *(const v8bf*)&Wth[g];
            *(v8bf*)&sBl[r * SB + c8] = *(const v8bf*)&Wtl[g];
        }
        __syncthreads();

        // ---- B fragments: lane = column n, K-runs 8h..8h+7 and 16+8h.. ----
        v16bf bhf[4], blf[4];
#pragma unroll
        for (int nt = 0; nt < 4; ++nt) {
            const __bf16* pB = &sBh[(wc + nt * 16 + lid) * SB + (hl << 3)];
            const __bf16* qB = &sBl[(wc + nt * 16 + lid) * SB + (hl << 3)];
            v8bf b0 = *(const v8bf*)pB;
            v8bf b1 = *(const v8bf*)(pB + 16);
            v8bf c0 = *(const v8bf*)qB;
            v8bf c1 = *(const v8bf*)(qB + 16);
            bhf[nt] = __builtin_shufflevector(b0, b1, 0,1,2,3,4,5,6,7,8,9,10,11,12,13,14,15);
            blf[nt] = __builtin_shufflevector(c0, c1, 0,1,2,3,4,5,6,7,8,9,10,11,12,13,14,15);
        }
        // ---- A fragments + WMMA -------------------------------------------
#pragma unroll
        for (int mt = 0; mt < 2; ++mt) {
            const __bf16* pA = &sAh[(wr + mt * 16 + lid) * SA + (hl << 3)];
            const __bf16* qA = &sAl[(wr + mt * 16 + lid) * SA + (hl << 3)];
            v8bf a0 = *(const v8bf*)pA;
            v8bf a1 = *(const v8bf*)(pA + 16);
            v8bf e0 = *(const v8bf*)qA;
            v8bf e1 = *(const v8bf*)(qA + 16);
            v16bf ah = __builtin_shufflevector(a0, a1, 0,1,2,3,4,5,6,7,8,9,10,11,12,13,14,15);
            v16bf al = __builtin_shufflevector(e0, e1, 0,1,2,3,4,5,6,7,8,9,10,11,12,13,14,15);
#pragma unroll
            for (int nt = 0; nt < 4; ++nt) {
                acc[mt][nt] = __builtin_amdgcn_wmma_f32_16x16x32_bf16(
                    false, ah, false, bhf[nt], (short)0, acc[mt][nt], false, false);
                acc[mt][nt] = __builtin_amdgcn_wmma_f32_16x16x32_bf16(
                    false, ah, false, blf[nt], (short)0, acc[mt][nt], false, false);
                acc[mt][nt] = __builtin_amdgcn_wmma_f32_16x16x32_bf16(
                    false, al, false, bhf[nt], (short)0, acc[mt][nt], false, false);
            }
        }
        __syncthreads();
    }

    // ---- epilogue: C/D layout -> element r: M = 8*half + r, N = lid -------
#pragma unroll
    for (int mt = 0; mt < 2; ++mt)
#pragma unroll
        for (int nt = 0; nt < 4; ++nt) {
            int gcol = n0 + wc + nt * 16 + lid;
            float bv = bias ? bias[gcol] : 0.0f;
#pragma unroll
            for (int r = 0; r < 8; ++r) {
                int grow = m0 + wr + mt * 16 + hl * 8 + r;
                float v = acc[mt][nt][r] + bv;
                if (relu) v = fmaxf(v, 0.0f);
                C[(size_t)grow * N + gcol] = v;
            }
        }
}

// ---------------------------------------------------------------------------
__global__ void zero_kernel(float4* __restrict__ p, int n4) {
    int i = blockIdx.x * 256 + threadIdx.x;
    if (i < n4) p[i] = make_float4(0.f, 0.f, 0.f, 0.f);
}

// ---------------------------------------------------------------------------
// qs score per (b,h,l): 0.5*||q||2 + 0.3*entropy(softmax(q)) + 0.2*var(q,ddof=1)
// ---------------------------------------------------------------------------
__global__ __launch_bounds__(256) void qs_kernel(const float* __restrict__ Q,
                                                 float* __restrict__ qs) {
    int idx = blockIdx.x * 256 + threadIdx.x;       // [0, MROWS*NHEAD)
    int m = idx >> 3;
    int h = idx & 7;
    const float* q = Q + (size_t)m * DMODEL + h * DK;
    float s1 = 0.f, s2 = 0.f, mx = -3.402823466e38f;
    for (int d = 0; d < DK; ++d) {
        float v = q[d];
        s1 += v; s2 += v * v; mx = fmaxf(mx, v);
    }
    float se = 0.f;
    for (int d = 0; d < DK; ++d) se += expf(q[d] - mx);
    float ent = 0.f;
    float inv = 1.0f / se;
    for (int d = 0; d < DK; ++d) {
        float p = expf(q[d] - mx) * inv;
        ent -= p * logf(p + 1e-9f);
    }
    float l2   = sqrtf(s2);
    float mean = s1 * (1.0f / DK);
    float var  = (s2 - DK * mean * mean) * (1.0f / (DK - 1));
    int b = m / SEQL, l = m % SEQL;
    qs[((size_t)(b * NHEAD + h)) * SEQL + l] = 0.5f * l2 + 0.3f * ent + 0.2f * var;
}

// ---------------------------------------------------------------------------
// Per-batch u = clip(round(std(qs[b,0,:],ddof=1)/(mean+1e-6)*10), 3, 10)
// ---------------------------------------------------------------------------
__global__ __launch_bounds__(256) void u_kernel(const float* __restrict__ qs,
                                                int* __restrict__ u) {
    int b = blockIdx.x;
    const float* v = qs + (size_t)(b * NHEAD) * SEQL;   // head 0
    __shared__ float r1[256], r2[256];
    float s1 = 0.f, s2 = 0.f;
    for (int i = 0; i < SEQL / 256; ++i) {
        float x = v[threadIdx.x + i * 256];
        s1 += x; s2 += x * x;
    }
    r1[threadIdx.x] = s1; r2[threadIdx.x] = s2;
    __syncthreads();
    for (int s = 128; s > 0; s >>= 1) {
        if (threadIdx.x < s) { r1[threadIdx.x] += r1[threadIdx.x + s];
                               r2[threadIdx.x] += r2[threadIdx.x + s]; }
        __syncthreads();
    }
    if (threadIdx.x == 0) {
        float mean = r1[0] / (float)SEQL;
        float var  = fmaxf(0.f, (r2[0] - SEQL * mean * mean)) / (float)(SEQL - 1);
        float f = rintf(sqrtf(var) / (mean + 1e-6f) * 10.0f);  // RNE == jnp.round
        f = fminf(fmaxf(f, 3.0f), 10.0f);
        u[b] = (int)f;
    }
}

// ---------------------------------------------------------------------------
// Top-10 of qs[b,h,:] (descending, ties -> lower index), per (b,h) workgroup.
// ---------------------------------------------------------------------------
__global__ __launch_bounds__(256) void topk_kernel(const float* __restrict__ qs,
                                                   int* __restrict__ top_idx) {
    int bh = blockIdx.x;
    const float* v = qs + (size_t)bh * SEQL;
    __shared__ float bv[256];
    __shared__ int   bi[256];
    __shared__ int   chosen[UMAX];
    int tid = threadIdx.x;
    for (int pass = 0; pass < UMAX; ++pass) {
        float best = -3.402823466e38f; int besti = SEQL;
        for (int li = 0; li < SEQL / 256; ++li) {
            int l = tid + li * 256;
            bool skip = false;
            for (int p = 0; p < pass; ++p) if (chosen[p] == l) skip = true;
            if (!skip) {
                float x = v[l];
                if (x > best || (x == best && l < besti)) { best = x; besti = l; }
            }
        }
        bv[tid] = best; bi[tid] = besti;
        __syncthreads();
        for (int s = 128; s > 0; s >>= 1) {
            if (tid < s) {
                if (bv[tid + s] > bv[tid] ||
                    (bv[tid + s] == bv[tid] && bi[tid + s] < bi[tid])) {
                    bv[tid] = bv[tid + s]; bi[tid] = bi[tid + s];
                }
            }
            __syncthreads();
        }
        if (tid == 0) { chosen[pass] = bi[0]; top_idx[bh * UMAX + pass] = bi[0]; }
        __syncthreads();
    }
}

// ---------------------------------------------------------------------------
// Sparse attention for one (b,h,slot j): softmax(Qs . K^T / 8) @ V, scattered
// into the (pre-zeroed) dense y buffer at row top_idx.
// ---------------------------------------------------------------------------
__global__ __launch_bounds__(256) void attn_kernel(const float* __restrict__ Q,
                                                   const float* __restrict__ Km,
                                                   const float* __restrict__ Vm,
                                                   const int* __restrict__ top_idx,
                                                   const int* __restrict__ u,
                                                   float* __restrict__ y) {
    int blk = blockIdx.x;                // ((b*NHEAD)+h)*UMAX + j
    int j  = blk % UMAX;
    int bh = blk / UMAX;
    int h  = bh % NHEAD;
    int b  = bh / NHEAD;
    if (j >= u[b]) return;               // uniform across block

    __shared__ float sc[SEQL];
    __shared__ float qrow[DK];
    __shared__ float red[256];
    __shared__ float part[4][DK];

    int tid  = threadIdx.x;
    int qidx = top_idx[bh * UMAX + j];
    if (tid < DK) qrow[tid] = Q[((size_t)(b * SEQL + qidx)) * DMODEL + h * DK + tid];
    __syncthreads();

    float lmax = -3.402823466e38f;
    for (int li = 0; li < SEQL / 256; ++li) {
        int l = tid + li * 256;
        const float* kp = Km + ((size_t)(b * SEQL + l)) * DMODEL + h * DK;
        float s = 0.f;
        for (int d = 0; d < DK; ++d) s += qrow[d] * kp[d];
        s *= 0.125f;                     // 1/sqrt(64)
        sc[l] = s;
        lmax = fmaxf(lmax, s);
    }
    red[tid] = lmax; __syncthreads();
    for (int s = 128; s > 0; s >>= 1) {
        if (tid < s) red[tid] = fmaxf(red[tid], red[tid + s]);
        __syncthreads();
    }
    float mx = red[0];
    __syncthreads();
    float lsum = 0.f;
    for (int li = 0; li < SEQL / 256; ++li) {
        int l = tid + li * 256;
        float e = expf(sc[l] - mx);
        sc[l] = e; lsum += e;
    }
    red[tid] = lsum; __syncthreads();
    for (int s = 128; s > 0; s >>= 1) {
        if (tid < s) red[tid] += red[tid + s];
        __syncthreads();
    }
    float inv = 1.0f / red[0];
    __syncthreads();

    int d = tid & (DK - 1), pt = tid >> 6;
    float acc = 0.f;
    const float* vb = Vm + ((size_t)(b * SEQL + pt * 1024)) * DMODEL + h * DK + d;
    for (int l = 0; l < 1024; ++l) acc += sc[pt * 1024 + l] * vb[(size_t)l * DMODEL];
    part[pt][d] = acc;
    __syncthreads();
    if (pt == 0) {
        float o = (part[0][d] + part[1][d] + part[2][d] + part[3][d]) * inv;
        y[((size_t)(b * SEQL + qidx)) * DMODEL + h * DK + d] = o;
    }
}

// ---------------------------------------------------------------------------
// In-place residual add + LayerNorm over last dim (512), one row per block.
// ---------------------------------------------------------------------------
__global__ __launch_bounds__(256) void add_ln_kernel(float* __restrict__ h,
                                                     const float* __restrict__ a,
                                                     const float* __restrict__ g,
                                                     const float* __restrict__ be) {
    int row = blockIdx.x, tid = threadIdx.x;
    size_t base = (size_t)row * DMODEL;
    float x0 = h[base + tid]       + a[base + tid];
    float x1 = h[base + tid + 256] + a[base + tid + 256];
    __shared__ float r1[256], r2[256];
    r1[tid] = x0 + x1; r2[tid] = x0 * x0 + x1 * x1;
    __syncthreads();
    for (int s = 128; s > 0; s >>= 1) {
        if (tid < s) { r1[tid] += r1[tid + s]; r2[tid] += r2[tid + s]; }
        __syncthreads();
    }
    float mean = r1[0] * (1.0f / DMODEL);
    float var  = r2[0] * (1.0f / DMODEL) - mean * mean;
    float rstd = rsqrtf(var + 1e-5f);
    h[base + tid]       = (x0 - mean) * rstd * g[tid]       + be[tid];
    h[base + tid + 256] = (x1 - mean) * rstd * g[tid + 256] + be[tid + 256];
}

// ---------------------------------------------------------------------------
// Final projection: pred[b,t] = h[b, L-96+t, :] . w_out + b_out
// ---------------------------------------------------------------------------
__global__ __launch_bounds__(256) void final_kernel(const float* __restrict__ h,
                                                    const float* __restrict__ w_out,
                                                    const float* __restrict__ b_out,
                                                    float* __restrict__ out) {
    int idx = blockIdx.x * 256 + threadIdx.x;   // < BATCH*OUTL
    if (idx >= BATCH * OUTL) return;
    int b = idx / OUTL, t = idx % OUTL;
    size_t row = (size_t)b * SEQL + (SEQL - OUTL) + t;
    const float* hr = h + row * DMODEL;
    float s = b_out[0];
    for (int d = 0; d < DMODEL; ++d) s += hr[d] * w_out[d];
    out[idx] = s;
}

// ---------------------------------------------------------------------------
extern "C" void kernel_launch(void* const* d_in, const int* in_sizes, int n_in,
                              void* d_out, int out_size, void* d_ws, size_t ws_size,
                              hipStream_t stream) {
    const float* x       = (const float*)d_in[0];
    const float* w_embed = (const float*)d_in[1];
    const float* b_embed = (const float*)d_in[2];
    const float* wq = (const float*)d_in[3];
    const float* bq = (const float*)d_in[4];
    const float* wk = (const float*)d_in[5];
    const float* bk = (const float*)d_in[6];
    const float* wv = (const float*)d_in[7];
    const float* bv = (const float*)d_in[8];
    const float* wo = (const float*)d_in[9];
    const float* bo = (const float*)d_in[10];
    const float* w1 = (const float*)d_in[11];
    const float* b1 = (const float*)d_in[12];
    const float* w2 = (const float*)d_in[13];
    const float* b2 = (const float*)d_in[14];
    const float* g1  = (const float*)d_in[15];
    const float* be1 = (const float*)d_in[16];
    const float* g2  = (const float*)d_in[17];
    const float* be2 = (const float*)d_in[18];
    const float* w_out = (const float*)d_in[19];
    const float* b_out = (const float*)d_in[20];

    // ---- workspace layout ----
    float* ws  = (float*)d_ws;
    float* h   = ws;                                   // MROWS * 512
    float* t1  = h  + (size_t)MROWS * DMODEL;          // MROWS * 512 (y / ffn2 out)
    float* big = t1 + (size_t)MROWS * DMODEL;          // MROWS * 2048 (Q|K|V / ffn hidden)
    float* qsb = big + (size_t)MROWS * DFF;            // B*H*L
    const size_t DD  = (size_t)DMODEL * DMODEL;        // 262144
    const size_t DDF = (size_t)DMODEL * DFF;           // 1048576
    __bf16* bfp = (__bf16*)(qsb + (size_t)BATCH * NHEAD * SEQL);
    __bf16 *wq_h = bfp,            *wq_l = wq_h + DD;
    __bf16 *wk_h = wq_l + DD,      *wk_l = wk_h + DD;
    __bf16 *wv_h = wk_l + DD,      *wv_l = wv_h + DD;
    __bf16 *wo_h = wv_l + DD,      *wo_l = wo_h + DD;
    __bf16 *w1_h = wo_l + DD,      *w1_l = w1_h + DDF;
    __bf16 *w2_h = w1_l + DDF,     *w2_l = w2_h + DDF;
    int* topi = (int*)(w2_l + DDF);                    // B*H*UMAX
    int* ub   = topi + BATCH * NHEAD * UMAX;           // B

    float* Qb = big;
    float* Kb = big + (size_t)MROWS * DMODEL;
    float* Vb = big + 2 * (size_t)MROWS * DMODEL;

    // ---- split weights (transposed) to bf16 hi/lo, once per launch ----
    split_wt_kernel<<<(int)((DD  + 255) / 256), 256, 0, stream>>>(wq, wq_h, wq_l, DMODEL, DMODEL);
    split_wt_kernel<<<(int)((DD  + 255) / 256), 256, 0, stream>>>(wk, wk_h, wk_l, DMODEL, DMODEL);
    split_wt_kernel<<<(int)((DD  + 255) / 256), 256, 0, stream>>>(wv, wv_h, wv_l, DMODEL, DMODEL);
    split_wt_kernel<<<(int)((DD  + 255) / 256), 256, 0, stream>>>(wo, wo_h, wo_l, DMODEL, DMODEL);
    split_wt_kernel<<<(int)((DDF + 255) / 256), 256, 0, stream>>>(w1, w1_h, w1_l, DMODEL, DFF);
    split_wt_kernel<<<(int)((DDF + 255) / 256), 256, 0, stream>>>(w2, w2_h, w2_l, DFF, DMODEL);

    // ---- embed ----
    embed_kernel<<<(MROWS * DMODEL) / 256, 256, 0, stream>>>(x, w_embed, b_embed, h);

    const dim3 gemmDD(DMODEL / BN, MROWS / BM);   // (4, 512)
    const dim3 gemmDF(DFF / BN,    MROWS / BM);   // (16, 512)

    for (int layer = 0; layer < 2; ++layer) {
        // Q/K/V projections
        gemm_bf16x3_kernel<<<gemmDD, 256, 0, stream>>>(h, wq_h, wq_l, bq, Qb, DMODEL, DMODEL, 0);
        gemm_bf16x3_kernel<<<gemmDD, 256, 0, stream>>>(h, wk_h, wk_l, bk, Kb, DMODEL, DMODEL, 0);
        gemm_bf16x3_kernel<<<gemmDD, 256, 0, stream>>>(h, wv_h, wv_l, bv, Vb, DMODEL, DMODEL, 0);

        // prob-sparse selection
        qs_kernel<<<(MROWS * NHEAD) / 256, 256, 0, stream>>>(Qb, qsb);
        u_kernel<<<BATCH, 256, 0, stream>>>(qsb, ub);
        topk_kernel<<<BATCH * NHEAD, 256, 0, stream>>>(qsb, topi);

        // sparse attention into dense y (t1)
        zero_kernel<<<(MROWS * DMODEL / 4 + 255) / 256, 256, 0, stream>>>((float4*)t1,
                                                                          MROWS * DMODEL / 4);
        attn_kernel<<<BATCH * NHEAD * UMAX, 256, 0, stream>>>(Qb, Kb, Vb, topi, ub, t1);

        // output projection (Q region reusable now), residual + LN1
        gemm_bf16x3_kernel<<<gemmDD, 256, 0, stream>>>(t1, wo_h, wo_l, bo, Qb, DMODEL, DMODEL, 0);
        add_ln_kernel<<<MROWS, 256, 0, stream>>>(h, Qb, g1, be1);

        // FFN: relu(h@w1+b1)@w2+b2, residual + LN2
        gemm_bf16x3_kernel<<<gemmDF, 256, 0, stream>>>(h, w1_h, w1_l, b1, big, DFF, DMODEL, 1);
        gemm_bf16x3_kernel<<<gemmDD, 256, 0, stream>>>(big, w2_h, w2_l, b2, t1, DMODEL, DFF, 0);
        add_ln_kernel<<<MROWS, 256, 0, stream>>>(h, t1, g2, be2);
    }

    // adaptive_decomp is an identity (seasonal + trend == h): skip straight to
    // the output projection on the last OUT_LEN rows.
    final_kernel<<<(BATCH * OUTL + 255) / 256, 256, 0, stream>>>(h, w_out, b_out,
                                                                 (float*)d_out);
    (void)in_sizes; (void)n_in; (void)out_size; (void)ws_size;
}
